// GCN_layer_88579405512833
// MI455X (gfx1250) — matrix operand order repaired and checked by reading, not compiled, not run on hardware
//
#include <hip/hip_runtime.h>
#include <hip/hip_bf16.h>

typedef __attribute__((ext_vector_type(16))) _Float16 v16h;
typedef __attribute__((ext_vector_type(8)))  _Float16 v8h;
typedef __attribute__((ext_vector_type(4)))  _Float16 v4h;
typedef __attribute__((ext_vector_type(8)))  float    v8f;

#define GCN_F 128
#define GCN_H 128

// ---------------------------------------------------------------- degree init
__global__ void init_deg_kernel(float* __restrict__ deg, int N) {
    int i = blockIdx.x * blockDim.x + threadIdx.x;
    if (i < N) deg[i] = 1.0f;  // self-loop contributes 1 to every node's degree
}

// ---------------------------------------------------------------- degree count
__global__ void count_deg_kernel(const long long* __restrict__ dst,
                                 float* __restrict__ deg, int E) {
    int e = blockIdx.x * blockDim.x + threadIdx.x;
    if (e < E) {
        int d = (int)dst[e];
        __hip_atomic_fetch_add(&deg[d], 1.0f, __ATOMIC_RELAXED,
                               __HIP_MEMORY_SCOPE_AGENT);
    }
}

// ---------------------------------------------------------------- dinv = rsqrt
__global__ void rsqrt_kernel(float* __restrict__ deg, int N) {
    int i = blockIdx.x * blockDim.x + threadIdx.x;
    if (i < N) {
        float d = deg[i];
        deg[i] = (d > 0.0f) ? __frsqrt_rn(d) : 0.0f;  // in place -> dinv
    }
}

// ---------------------------------------------------------------- x -> f16
__global__ void convert_x_kernel(const float* __restrict__ x,
                                 _Float16* __restrict__ xh, long long n4) {
    long long i = (long long)blockIdx.x * blockDim.x + threadIdx.x;
    if (i < n4) {
        float4 v = ((const float4*)x)[i];
        v4h o;
        o.x = (_Float16)v.x; o.y = (_Float16)v.y;
        o.z = (_Float16)v.z; o.w = (_Float16)v.w;
        ((v4h*)xh)[i] = o;
    }
}

// ------------------------------------------------- W[k][n] -> Wt[n][k] (f16)
__global__ void transpose_w_kernel(const float* __restrict__ W,
                                   _Float16* __restrict__ wt) {
    int idx = blockIdx.x * blockDim.x + threadIdx.x;  // 128*128
    int k = idx >> 7;
    int n = idx & 127;
    wt[n * GCN_F + k] = (_Float16)W[k * GCN_H + n];
}

// ---------------------------------------------------------------- WMMA GEMM
// h[N x 128] = xh[N x 128] @ Wt^T   (Wt is [n][k] so lanes read contiguous K)
// One block = 8 waves = 128 rows. Wt (32 KB) is staged once per block into
// LDS; each wave computes a 16-row stripe across all 8 column tiles, keeping
// the A fragment in registers. Per k-step: batch-load all 8 B fragments from
// LDS, then fire 8 back-to-back v_wmma ops (no per-WMMA VMEM wait).
__global__ __launch_bounds__(256) void gemm_wmma_kernel(
    const _Float16* __restrict__ xh, const _Float16* __restrict__ wt,
    float* __restrict__ h, int N) {
    __shared__ _Float16 wls[GCN_F * GCN_H];  // 32 KB of the 320 KB WGP LDS

    const int tid  = threadIdx.x;
    const int lane = tid & 31;
    const int wave = tid >> 5;

    // cooperative fill: 256 threads x 8 iters x 16B = 32 KB
#pragma unroll
    for (int i = 0; i < 8; ++i) {
        int idx = (i * 256 + tid) * 8;  // in halves
        *(v8h*)(wls + idx) = *(const v8h*)(wt + idx);
    }
    __syncthreads();

    const int row0 = (blockIdx.x * 8 + wave) * 16;
    if (row0 >= N) return;

    const int lo = lane & 15;  // M for A, N for B/C/D
    const int hi = lane >> 4;  // K-half selector

    v8f acc[8] = {};

    const _Float16* arow = xh + (size_t)(row0 + lo) * GCN_F;

    union Frag { v16h v; v8h half8[2]; };

#pragma unroll
    for (int kk = 0; kk < GCN_F; kk += 32) {
        // A fragment: 16-bit 16x32 layout — lanes 0-15: K 0..7 / 16..23,
        // lanes 16-31: K 8..15 / 24..31 (per ISA 7.12.2)
        Frag a;
        const _Float16* ap = arow + kk + 8 * hi;
        a.half8[0] = *(const v8h*)(ap);
        a.half8[1] = *(const v8h*)(ap + 16);

        // Batch all 8 column-tile B fragments from LDS (lanes 0-15 hold
        // K=0..7 / 16..23, lanes 16-31 K=8..15 / 24..31; lane%16 selects column n)
        Frag bf[8];
#pragma unroll
        for (int ct = 0; ct < 8; ++ct) {
            const _Float16* bp = wls + (ct * 16 + lo) * GCN_F + kk + 8 * hi;
            bf[ct].half8[0] = *(const v8h*)(bp);
            bf[ct].half8[1] = *(const v8h*)(bp + 16);
        }

        // 8 back-to-back WMMAs on independent accumulators
#pragma unroll
        for (int ct = 0; ct < 8; ++ct) {
            acc[ct] = __builtin_amdgcn_wmma_f32_16x16x32_f16(
                false, a.v, false, bf[ct].v, (short)0, acc[ct], false, false);
        }
    }

    // D layout: VGPR r holds row (r + 8*hi), column = ct*16 + lo
#pragma unroll
    for (int ct = 0; ct < 8; ++ct) {
        float* op = h + (size_t)(row0 + 8 * hi) * GCN_H + ct * 16 + lo;
#pragma unroll
        for (int r = 0; r < 8; ++r) {
            op[(size_t)r * GCN_H] = acc[ct][r];
        }
    }
}

// --------------------------------------- out = dinv[i]^2 * h[i] + b  (no atomics)
__global__ void self_loop_bias_kernel(const float* __restrict__ h,
                                      const float* __restrict__ dinv,
                                      const float* __restrict__ b,
                                      float* __restrict__ out, int N) {
    long long idx = (long long)blockIdx.x * blockDim.x + threadIdx.x;  // N*32
    if (idx >= (long long)N * (GCN_H / 4)) return;
    int i = (int)(idx >> 5);
    int jv = (int)(idx & 31);
    float di = dinv[i];
    float nrm = di * di;
    float4 hv = ((const float4*)(h + (size_t)i * GCN_H))[jv];
    float4 bv = ((const float4*)b)[jv];
    float4 o;
    o.x = nrm * hv.x + bv.x;
    o.y = nrm * hv.y + bv.y;
    o.z = nrm * hv.z + bv.z;
    o.w = nrm * hv.w + bv.w;
    ((float4*)(out + (size_t)i * GCN_H))[jv] = o;
}

// ---------------------------------------------------------------- edge scatter
// One wave32 per edge. Lane l handles features [4l, 4l+4): coalesced 512B
// row gather from h (L2-resident) + 4 relaxed f32 atomics into out (L2).
__global__ __launch_bounds__(256) void edge_scatter_kernel(
    const long long* __restrict__ src, const long long* __restrict__ dst,
    const float* __restrict__ dinv, const float* __restrict__ h,
    float* __restrict__ out, int E) {
    int e = (int)(((long long)blockIdx.x * blockDim.x + threadIdx.x) >> 5);
    int lane = threadIdx.x & 31;
    if (e >= E) return;
    int s = (int)src[e];
    int d = (int)dst[e];
    float nrm = dinv[s] * dinv[d];
    float4 v = ((const float4*)(h + (size_t)s * GCN_H))[lane];
    float* o = out + (size_t)d * GCN_H + lane * 4;
    __hip_atomic_fetch_add(o + 0, nrm * v.x, __ATOMIC_RELAXED, __HIP_MEMORY_SCOPE_AGENT);
    __hip_atomic_fetch_add(o + 1, nrm * v.y, __ATOMIC_RELAXED, __HIP_MEMORY_SCOPE_AGENT);
    __hip_atomic_fetch_add(o + 2, nrm * v.z, __ATOMIC_RELAXED, __HIP_MEMORY_SCOPE_AGENT);
    __hip_atomic_fetch_add(o + 3, nrm * v.w, __ATOMIC_RELAXED, __HIP_MEMORY_SCOPE_AGENT);
}

// ---------------------------------------------------------------- launch
extern "C" void kernel_launch(void* const* d_in, const int* in_sizes, int n_in,
                              void* d_out, int out_size, void* d_ws, size_t ws_size,
                              hipStream_t stream) {
    const float* x = (const float*)d_in[0];
    const long long* edge_index = (const long long*)d_in[1];  // int64 per reference
    const float* W = (const float*)d_in[2];
    const float* b = (const float*)d_in[3];
    float* out = (float*)d_out;

    const int N = in_sizes[0] / GCN_F;
    const int E = in_sizes[1] / 2;
    const long long* src = edge_index;
    const long long* dst = edge_index + E;

    // workspace layout (all offsets 256B aligned for N=100000)
    char* ws = (char*)d_ws;
    float* h = (float*)ws;                                     // N*128 f32
    size_t off = (size_t)N * GCN_H * sizeof(float);
    _Float16* xh = (_Float16*)(ws + off);                      // N*128 f16
    off += (size_t)N * GCN_F * sizeof(_Float16);
    _Float16* wt = (_Float16*)(ws + off);                      // 128*128 f16
    off += (size_t)GCN_F * GCN_H * sizeof(_Float16);
    float* deg = (float*)(ws + off);                           // N f32 (deg->dinv)

    const int T = 256;

    init_deg_kernel<<<(N + T - 1) / T, T, 0, stream>>>(deg, N);
    count_deg_kernel<<<(E + T - 1) / T, T, 0, stream>>>(dst, deg, E);
    rsqrt_kernel<<<(N + T - 1) / T, T, 0, stream>>>(deg, N);

    long long n4 = (long long)N * GCN_F / 4;
    convert_x_kernel<<<(int)((n4 + T - 1) / T), T, 0, stream>>>(x, xh, n4);
    transpose_w_kernel<<<(GCN_F * GCN_H) / T, T, 0, stream>>>(W, wt);

    gemm_wmma_kernel<<<(N + 127) / 128, T, 0, stream>>>(xh, wt, h, N);

    long long nsl = (long long)N * (GCN_H / 4);
    self_loop_bias_kernel<<<(int)((nsl + T - 1) / T), T, 0, stream>>>(h, deg, b, out, N);

    long long ethreads = (long long)E * 32;
    edge_scatter_kernel<<<(int)((ethreads + T - 1) / T), T, 0, stream>>>(
        src, dst, deg, h, out, E);
}
